// GCMC_62397284876369
// MI455X (gfx1250) — matrix operand order repaired and checked by reading, not compiled, not run on hardware
//
#include <hip/hip_runtime.h>
#include <hip/hip_bf16.h>

// ---------------------------------------------------------------------------
// Types / constants
// ---------------------------------------------------------------------------
typedef unsigned short u16;
typedef unsigned int   u32;
typedef __attribute__((ext_vector_type(16))) __bf16 bf16x16;
typedef __attribute__((ext_vector_type(8)))  float  f32x8;

#define GN    4096          // N nodes
#define GIN   4096          // input features
#define GH    512           // hidden
#define GH2   1024          // 2*H
#define GEH   1024          // EXP*H
#define GOUT  256
#define GE    262144        // edges
#define GDS   16
#define GDCONV 4
#define GGRIDS 16
#define GNUMR 2048
#define LN_EPS 1e-5f

// GEMM epilogue flags
#define GF_BIAS     1
#define GF_ACCUM    2
#define GF_RELU     4
#define GF_SIGMOID  8
#define GF_DIVSCALE 16

__device__ __forceinline__ u16 f2bf(float f) {
    u32 u = __float_as_uint(f);
    u32 r = u + 0x7FFFu + ((u >> 16) & 1u);   // round-to-nearest-even
    return (u16)(r >> 16);
}
__device__ __forceinline__ float sigmoidf_(float x) { return 1.0f / (1.0f + expf(-x)); }

// ---------------------------------------------------------------------------
// WMMA bf16 GEMM:  C[M,N] = epilogue( A[M,K] @ B[K,N] ),  B passed TRANSPOSED
// A row-major [M,K] bf16; Bt row-major [N,K] bf16 (i.e. B column-major);
// C f32. M%128==0, N%64==0, K%64==0.
// 256 threads = 8 waves (4x2); block tile 128x64; wave tile 32x32 (4 accs).
// A staged through LDS; B fragments (16 contiguous K per lane) loaded straight
// from global. K-loop unrolled x2 with two NAMED fragment sets (b0/b1) so the
// ping-pong is statically indexed and stays entirely in VGPRs.
// ---------------------------------------------------------------------------
#define TM 128
#define TN 64
#define TK 32

union BU { bf16x16 v; uint4 q[2]; };

__global__ __launch_bounds__(256)
void k_gemm_bf16(const u16* __restrict__ A, const u16* __restrict__ Bt,
                 const float* __restrict__ bias, float* __restrict__ C,
                 int M, int N, int K, int flags, const float* __restrict__ scale_ptr)
{
    __shared__ u16 sA[TM * TK];   // 8 KB, row-major [row][k]

    const int tid  = threadIdx.x;
    const int lane = tid & 31;
    const int wave = tid >> 5;        // 0..7
    const int mt   = wave & 3;        // 32-row band
    const int nt   = wave >> 2;       // 32-col band
    const int bm   = blockIdx.y * TM;
    const int bn   = blockIdx.x * TN;

    const int half = lane >> 4;       // 0/1
    const int l16  = lane & 15;

    // A staging assignments (two b128 chunks per thread)
    const int ar0 = tid >> 2,          aq0 = tid & 3;          // rows 0..63
    const int ar1 = (tid + 256) >> 2,  aq1 = (tid + 256) & 3;  // rows 64..127

    const u16* gA0 = A + (size_t)(bm + ar0) * K + aq0 * 8;
    const u16* gA1 = A + (size_t)(bm + ar1) * K + aq1 * 8;
    // per-lane B base: column (bn + nt*32 + t*16 + l16), contiguous K from half*16
    const u16* gB  = Bt + (size_t)(bn + nt * 32 + l16) * K + half * 16;

    f32x8 acc[2][2] = {};

    // B fragment layout: vgpr j holds K = half*16 + 2j (+1) -> 16 contiguous bf16
    BU b0[2], b1[2];
#pragma unroll
    for (int t = 0; t < 2; ++t) {
        const u16* bp = gB + (size_t)(t * 16) * K;
        b0[t].q[0] = *reinterpret_cast<const uint4*>(bp);
        b0[t].q[1] = *reinterpret_cast<const uint4*>(bp + 8);
    }
    uint4 ra0 = *reinterpret_cast<const uint4*>(gA0);
    uint4 ra1 = *reinterpret_cast<const uint4*>(gA1);

    for (int k0 = 0; k0 < K; k0 += 2 * TK) {
        // ================= phase 0 (uses b0, preloads b1) =================
        *reinterpret_cast<uint4*>(&sA[ar0 * TK + aq0 * 8]) = ra0;
        *reinterpret_cast<uint4*>(&sA[ar1 * TK + aq1 * 8]) = ra1;
        __syncthreads();

        // k0+TK is always in range (K % 64 == 0)
        ra0 = *reinterpret_cast<const uint4*>(gA0 + k0 + TK);
        ra1 = *reinterpret_cast<const uint4*>(gA1 + k0 + TK);
#pragma unroll
        for (int t = 0; t < 2; ++t) {
            const u16* bp = gB + (size_t)(t * 16) * K + k0 + TK;
            b1[t].q[0] = *reinterpret_cast<const uint4*>(bp);
            b1[t].q[1] = *reinterpret_cast<const uint4*>(bp + 8);
        }

        {
            // A fragment: u[0..7] contiguous at K=half*16 (matches B)
            union { bf16x16 v; uint4 q[2]; } af[2];
#pragma unroll
            for (int s = 0; s < 2; ++s) {
                const u16* ap = &sA[(mt * 32 + s * 16 + l16) * TK];
                af[s].q[0] = *reinterpret_cast<const uint4*>(ap + half * 16);
                af[s].q[1] = *reinterpret_cast<const uint4*>(ap + half * 16 + 8);
            }
#pragma unroll
            for (int s = 0; s < 2; ++s)
#pragma unroll
                for (int t = 0; t < 2; ++t)
                    acc[s][t] = __builtin_amdgcn_wmma_f32_16x16x32_bf16(
                        false, af[s].v, false, b0[t].v, (short)0, acc[s][t], false, false);
        }
        __syncthreads();

        // ================= phase 1 (uses b1, preloads b0) =================
        *reinterpret_cast<uint4*>(&sA[ar0 * TK + aq0 * 8]) = ra0;
        *reinterpret_cast<uint4*>(&sA[ar1 * TK + aq1 * 8]) = ra1;
        __syncthreads();

        if (k0 + 2 * TK < K) {
            ra0 = *reinterpret_cast<const uint4*>(gA0 + k0 + 2 * TK);
            ra1 = *reinterpret_cast<const uint4*>(gA1 + k0 + 2 * TK);
#pragma unroll
            for (int t = 0; t < 2; ++t) {
                const u16* bp = gB + (size_t)(t * 16) * K + k0 + 2 * TK;
                b0[t].q[0] = *reinterpret_cast<const uint4*>(bp);
                b0[t].q[1] = *reinterpret_cast<const uint4*>(bp + 8);
            }
            if (k0 + 3 * TK < K) {  // gfx1250 global_prefetch_b8
                __builtin_prefetch(gA0 + k0 + 3 * TK, 0, 3);
                __builtin_prefetch(gB + k0 + 3 * TK, 0, 3);
            }
        }

        {
            union { bf16x16 v; uint4 q[2]; } af[2];
#pragma unroll
            for (int s = 0; s < 2; ++s) {
                const u16* ap = &sA[(mt * 32 + s * 16 + l16) * TK];
                af[s].q[0] = *reinterpret_cast<const uint4*>(ap + half * 16);
                af[s].q[1] = *reinterpret_cast<const uint4*>(ap + half * 16 + 8);
            }
#pragma unroll
            for (int s = 0; s < 2; ++s)
#pragma unroll
                for (int t = 0; t < 2; ++t)
                    acc[s][t] = __builtin_amdgcn_wmma_f32_16x16x32_bf16(
                        false, af[s].v, false, b1[t].v, (short)0, acc[s][t], false, false);
        }
        __syncthreads();
    }

    // ---- epilogue; C layout: vgpr v -> row = v + 8*half, col = lane%16 ----
    float inv_scale = (flags & GF_DIVSCALE) ? (1.0f / scale_ptr[0]) : 1.0f;
#pragma unroll
    for (int s = 0; s < 2; ++s) {
#pragma unroll
        for (int t = 0; t < 2; ++t) {
#pragma unroll
            for (int v = 0; v < 8; ++v) {
                int grow = bm + mt * 32 + s * 16 + v + half * 8;
                int gcol = bn + nt * 32 + t * 16 + l16;
                size_t i0 = (size_t)grow * N + gcol;
                float r = acc[s][t][v];
                if (flags & GF_ACCUM)    r += C[i0];
                if (flags & GF_BIAS)     r += bias[gcol];
                if (flags & GF_DIVSCALE) r *= inv_scale;
                if (flags & GF_RELU)     r = fmaxf(r, 0.f);
                if (flags & GF_SIGMOID)  r = sigmoidf_(r);
                C[i0] = r;
            }
        }
    }
}

// ---------------------------------------------------------------------------
// Support kernels
// ---------------------------------------------------------------------------
__global__ void k_fillf(float* p, float v, size_t n) {
    size_t i = (size_t)blockIdx.x * blockDim.x + threadIdx.x;
    if (i < n) p[i] = v;
}
__global__ void k_copyf(float* d, const float* s, size_t n) {
    size_t i = (size_t)blockIdx.x * blockDim.x + threadIdx.x;
    if (i < n) d[i] = s[i];
}
__global__ void k_cast_bf16(const float* __restrict__ s, u16* __restrict__ d, size_t n) {
    size_t i = (size_t)blockIdx.x * blockDim.x + threadIdx.x;
    if (i < n) d[i] = f2bf(s[i]);
}
// transpose cast: src [R,C] f32 -> dst [C,R] bf16
__global__ void k_cast_bf16_t(const float* __restrict__ s, u16* __restrict__ d, int R, int Cc) {
    size_t i = (size_t)blockIdx.x * blockDim.x + threadIdx.x;
    if (i < (size_t)R * Cc) {
        int r = (int)(i / Cc), c = (int)(i % Cc);
        d[(size_t)c * R + r] = f2bf(s[i]);
    }
}
__global__ void k_scatter_edge(float* adj, const int* idx, const float* attr, int E, int N) {
    int e = blockIdx.x * blockDim.x + threadIdx.x;
    if (e < E) adj[(size_t)idx[e] * N + idx[E + e]] = attr[e];
}
__global__ void k_add_eye(float* adj, int N) {
    int i = blockIdx.x * blockDim.x + threadIdx.x;
    if (i < N) adj[(size_t)i * N + i] += 1.0f;
}
// column sums of adj (block per column)
__global__ void k_colsum(const float* __restrict__ adj, float* cs, int N) {
    __shared__ float red[256];
    int c = blockIdx.x;
    float s = 0.f;
    for (int r = threadIdx.x; r < N; r += 256) s += adj[(size_t)r * N + c];
    red[threadIdx.x] = s; __syncthreads();
    for (int k = 128; k > 0; k >>= 1) { if (threadIdx.x < k) red[threadIdx.x] += red[threadIdx.x + k]; __syncthreads(); }
    if (threadIdx.x == 0) cs[c] = red[0];
}
// Ahat = D^-1/2 A D^-1/2 cast to bf16
__global__ void k_norm_adj(const float* __restrict__ adj, const float* __restrict__ cs,
                           u16* __restrict__ out, int N) {
    size_t i = (size_t)blockIdx.x * blockDim.x + threadIdx.x;
    if (i < (size_t)N * N) {
        int r = (int)(i / N), c = (int)(i % N);
        out[i] = f2bf(adj[i] * rsqrtf(cs[r]) * rsqrtf(cs[c]));
    }
}
// out = g*a + (1-g)*b (optional relu)
__global__ void k_gate_combine(float* out, const float* g, const float* a, const float* b,
                               size_t n, int relu) {
    size_t i = (size_t)blockIdx.x * blockDim.x + threadIdx.x;
    if (i < n) {
        float r = g[i] * a[i] + (1.0f - g[i]) * b[i];
        out[i] = relu ? fmaxf(r, 0.f) : r;
    }
}
__global__ void k_deg(float* deg, const int* idx, int E) {
    int e = blockIdx.x * blockDim.x + threadIdx.x;
    if (e < E) atomicAdd(&deg[idx[E + e]], 1.0f);
}
__global__ void k_clampdeg(float* deg, int N) {
    int i = blockIdx.x * blockDim.x + threadIdx.x;
    if (i < N) deg[i] = fmaxf(deg[i], 1.0f);
}
// agg[col] += x[row]  (block per edge, 128 threads x 4 feats)
__global__ void k_agg(float* __restrict__ agg, const float* __restrict__ x, const int* idx, int E) {
    int e = blockIdx.x;
    int row = idx[e], cn = idx[E + e];
    for (int c = threadIdx.x; c < GH; c += 128)
        atomicAdd(&agg[(size_t)cn * GH + c], x[(size_t)row * GH + c]);
}
__global__ void k_aggdiv_bf16(const float* agg, const float* deg, u16* out, size_t n) {
    size_t i = (size_t)blockIdx.x * blockDim.x + threadIdx.x;
    if (i < n) out[i] = f2bf(agg[i] / deg[i / GH]);
}
// row layernorm (block per row)
__global__ void k_layernorm(const float* __restrict__ x, const float* g, const float* b,
                            float* __restrict__ out, int C) {
    __shared__ float rs[256], rq[256];
    int row = blockIdx.x;
    const float* xr = x + (size_t)row * C;
    float s = 0.f, q = 0.f;
    for (int c = threadIdx.x; c < C; c += 256) { float v = xr[c]; s += v; q += v * v; }
    rs[threadIdx.x] = s; rq[threadIdx.x] = q; __syncthreads();
    for (int k = 128; k > 0; k >>= 1) {
        if (threadIdx.x < k) { rs[threadIdx.x] += rs[threadIdx.x + k]; rq[threadIdx.x] += rq[threadIdx.x + k]; }
        __syncthreads();
    }
    float m = rs[0] / C;
    float var = rq[0] / C - m * m;
    float inv = rsqrtf(var + LN_EPS);
    for (int c = threadIdx.x; c < C; c += 256)
        out[(size_t)row * C + c] = (xr[c] - m) * inv * g[c] + b[c];
}
// batchnorm column stats (block per column, Nrows rows)
__global__ void k_colstats(const float* __restrict__ x, float* mean, float* rstd, int Nr, int C) {
    __shared__ float rs[256], rq[256];
    int c = blockIdx.x;
    float s = 0.f, q = 0.f;
    for (int r = threadIdx.x; r < Nr; r += 256) { float v = x[(size_t)r * C + c]; s += v; q += v * v; }
    rs[threadIdx.x] = s; rq[threadIdx.x] = q; __syncthreads();
    for (int k = 128; k > 0; k >>= 1) {
        if (threadIdx.x < k) { rs[threadIdx.x] += rs[threadIdx.x + k]; rq[threadIdx.x] += rq[threadIdx.x + k]; }
        __syncthreads();
    }
    if (threadIdx.x == 0) {
        float m = rs[0] / Nr;
        mean[c] = m;
        rstd[c] = rsqrtf(rq[0] / Nr - m * m + LN_EPS);
    }
}
__global__ void k_bn_apply(const float* x, const float* mean, const float* rstd,
                           const float* g, const float* b, float* out, size_t n, int C, int relu) {
    size_t i = (size_t)blockIdx.x * blockDim.x + threadIdx.x;
    if (i < n) {
        int c = (int)(i % C);
        float r = (x[i] - mean[c]) * rstd[c] * g[c] + b[c];
        out[i] = relu ? fmaxf(r, 0.f) : r;
    }
}
// depthwise causal conv: out[n,d] = b[d] + sum_k w[d*4+k]*h[n-3+k,d]
__global__ void k_dwconv(const float* __restrict__ h, const float* __restrict__ w,
                         const float* __restrict__ b, float* __restrict__ out, int Nn, int D) {
    size_t i = (size_t)blockIdx.x * blockDim.x + threadIdx.x;
    if (i < (size_t)Nn * D) {
        int n = (int)(i / D), d = (int)(i % D);
        float acc = b[d];
#pragma unroll
        for (int k = 0; k < GDCONV; ++k) {
            int nn = n - (GDCONV - 1) + k;
            if (nn >= 0) acc += w[d * GDCONV + k] * h[(size_t)nn * D + d];
        }
        out[i] = acc;
    }
}
__global__ void k_bc(const float* B, const float* C, float* bc, int D) {
    int d = blockIdx.x * blockDim.x + threadIdx.x;
    if (d < D) {
        float s = 0.f;
#pragma unroll
        for (int k = 0; k < GDS; ++k) s += B[d * GDS + k] * C[d * GDS + k];
        bc[d] = s;
    }
}
__global__ void k_ssm(const float* h, const float* bc, const float* D, float* y, size_t n) {
    size_t i = (size_t)blockIdx.x * blockDim.x + threadIdx.x;
    if (i < n) {
        int d = (int)(i % GEH);
        float v = h[i];
        y[i] = v * v * bc[d] + D[d] * v;
    }
}
// row L2 normalize (block per row)
__global__ void k_rownorm(const float* x, float* out, int C) {
    __shared__ float rs[256];
    int row = blockIdx.x;
    float s = 0.f;
    for (int c = threadIdx.x; c < C; c += 256) { float v = x[(size_t)row * C + c]; s += v * v; }
    rs[threadIdx.x] = s; __syncthreads();
    for (int k = 128; k > 0; k >>= 1) { if (threadIdx.x < k) rs[threadIdx.x] += rs[threadIdx.x + k]; __syncthreads(); }
    float inv = rsqrtf(rs[0]);
    for (int c = threadIdx.x; c < C; c += 256)
        out[(size_t)row * C + c] = x[(size_t)row * C + c] * inv;
}
// -log_softmax diagonal term per row / per column, scaled into accumulator
__global__ void k_lsm_diag_row(const float* __restrict__ lg, float* acc, int N) {
    __shared__ float rr[256];
    int i = blockIdx.x;
    const float* r = lg + (size_t)i * N;
    float mx = -3.4e38f;
    for (int c = threadIdx.x; c < N; c += 256) mx = fmaxf(mx, r[c]);
    rr[threadIdx.x] = mx; __syncthreads();
    for (int k = 128; k > 0; k >>= 1) { if (threadIdx.x < k) rr[threadIdx.x] = fmaxf(rr[threadIdx.x], rr[threadIdx.x + k]); __syncthreads(); }
    mx = rr[0]; __syncthreads();
    float se = 0.f;
    for (int c = threadIdx.x; c < N; c += 256) se += expf(r[c] - mx);
    rr[threadIdx.x] = se; __syncthreads();
    for (int k = 128; k > 0; k >>= 1) { if (threadIdx.x < k) rr[threadIdx.x] += rr[threadIdx.x + k]; __syncthreads(); }
    if (threadIdx.x == 0) {
        float loss = -(r[i] - mx - logf(rr[0]));
        atomicAdd(acc, loss / (2.0f * N));
    }
}
__global__ void k_lsm_diag_col(const float* __restrict__ lg, float* acc, int N) {
    __shared__ float rr[256];
    int i = blockIdx.x;   // column
    float mx = -3.4e38f;
    for (int r = threadIdx.x; r < N; r += 256) mx = fmaxf(mx, lg[(size_t)r * N + i]);
    rr[threadIdx.x] = mx; __syncthreads();
    for (int k = 128; k > 0; k >>= 1) { if (threadIdx.x < k) rr[threadIdx.x] = fmaxf(rr[threadIdx.x], rr[threadIdx.x + k]); __syncthreads(); }
    mx = rr[0]; __syncthreads();
    float se = 0.f;
    for (int r = threadIdx.x; r < N; r += 256) se += expf(lg[(size_t)r * N + i] - mx);
    rr[threadIdx.x] = se; __syncthreads();
    for (int k = 128; k > 0; k >>= 1) { if (threadIdx.x < k) rr[threadIdx.x] += rr[threadIdx.x + k]; __syncthreads(); }
    if (threadIdx.x == 0) {
        float loss = -(lg[(size_t)i * N + i] - mx - logf(rr[0]));
        atomicAdd(acc, loss / (2.0f * N));
    }
}
__global__ void k_fusion_w(const float* dw, float* fw) {
    if (threadIdx.x == 0 && blockIdx.x == 0) {
        float m = fmaxf(dw[0], dw[1]);
        float e0 = expf(dw[0] - m), e1 = expf(dw[1] - m);
        fw[0] = e0 / (e0 + e1); fw[1] = e1 / (e0 + e1);
    }
}
__global__ void k_fuse(const float* xg, const float* xm, const float* g, const float* fw,
                       float* xf, size_t n) {
    size_t i = (size_t)blockIdx.x * blockDim.x + threadIdx.x;
    if (i < n) {
        float base = fw[0] * xg[i] + fw[1] * xm[i];
        xf[i] = g[i] * base + (1.0f - g[i]) * xg[i];
    }
}
// RBF basis, written directly as bf16: [N, H*GRIDS]
__global__ void k_basis(const float* __restrict__ xl, const float* __restrict__ grid,
                        u16* __restrict__ out, int Nn) {
    size_t i = (size_t)blockIdx.x * blockDim.x + threadIdx.x;   // over N*H
    if (i < (size_t)Nn * GH) {
        float v = xl[i];
        const float inv_denom = (GGRIDS - 1) / 5.0f;   // 1/((2.5-(-2.5))/(G-1)) = 3
        size_t o = i * GGRIDS;
#pragma unroll
        for (int g = 0; g < GGRIDS; ++g) {
            float t = (v - grid[g]) * inv_denom;
            out[o + g] = f2bf(expf(-t * t));
        }
    }
}
__global__ void k_silu_bf16(const float* x, u16* out, size_t n) {
    size_t i = (size_t)blockIdx.x * blockDim.x + threadIdx.x;
    if (i < n) { float v = x[i]; out[i] = f2bf(v * sigmoidf_(v)); }
}
__global__ void k_colmean(const float* x, float* mean, int Nr, int C) {
    __shared__ float rs[256];
    int c = blockIdx.x;
    float s = 0.f;
    for (int r = threadIdx.x; r < Nr; r += 256) s += x[(size_t)r * C + c];
    rs[threadIdx.x] = s; __syncthreads();
    for (int k = 128; k > 0; k >>= 1) { if (threadIdx.x < k) rs[threadIdx.x] += rs[threadIdx.x + k]; __syncthreads(); }
    if (threadIdx.x == 0) mean[c] = rs[0] / Nr;
}
__global__ void k_submean(float* x, const float* mean, size_t n, int C) {
    size_t i = (size_t)blockIdx.x * blockDim.x + threadIdx.x;
    if (i < n) x[i] -= mean[i % C];
}
__global__ void k_put_scalar(float* dst, const float* src) {
    if (threadIdx.x == 0 && blockIdx.x == 0) dst[0] = src[0];
}

// ---------------------------------------------------------------------------
// Host orchestration
// ---------------------------------------------------------------------------
static inline dim3 ew(size_t n) { return dim3((unsigned)((n + 255) / 256)); }

static void gemm(hipStream_t s, const u16* A, const u16* Bt, const float* bias, float* C,
                 int M, int N, int K, int flags, const float* scale_ptr) {
    dim3 grid(N / TN, M / TM);
    k_gemm_bf16<<<grid, 256, 0, s>>>(A, Bt, bias, C, M, N, K, flags, scale_ptr);
}

extern "C" void kernel_launch(void* const* d_in, const int* in_sizes, int n_in,
                              void* d_out, int out_size, void* d_ws, size_t ws_size,
                              hipStream_t stream) {
    auto F = [&](int i) { return (const float*)d_in[i]; };
    const float* x_in    = F(0);
    const float* adj_in  = F(1);
    const int*   idx     = (const int*)d_in[2];
    const float* attr    = F(3);
    const float* gcn0    = F(4);
    const float* gcn1    = F(5);
    const float* gate_w  = F(6);  const float* gate_b = F(7);
    const float* res_w   = F(8);  const float* res_b  = F(9);
    const float* lin_w   = F(10); const float* lin_b  = F(11);
    // mamba layer base: 12 + 16*L (in_proj w/b, out_proj w/b, conv_w, conv_b, A,B,C,D,
    //                              graph_proj w/b, ln_g, ln_b, gate_proj w/b)
    const float* l1_w = F(44); const float* l1_b = F(45);
    const float* bn1g = F(46); const float* bn1b = F(47);
    const float* l2_w = F(48); const float* l2_b = F(49);
    const float* bn2g = F(50); const float* bn2b = F(51);
    const float* fus_w = F(52); const float* fus_b = F(53); const float* fus_dw = F(54);
    const float* kan_g = F(55); const float* kan_b = F(56); const float* kan_grid = F(57);
    const float* spline_w = F(58);
    const float* base_w = F(59); const float* base_b = F(60);
    const float* dec_w = F(61);
    const float* temp  = F(62);

    // ---- workspace carve ----
    char* wsp = (char*)d_ws;
    size_t off = 0;
    auto carve = [&](size_t bytes) -> void* {
        void* p = wsp + off;
        off += (bytes + 255) & ~(size_t)255;
        return p;
    };
    u16* Ahat   = (u16*)carve((size_t)GN * GN * 2);
    u16* bigbf  = (u16*)carve((size_t)GN * GIN * 2);       // x / adj cast
    u16* basisb = (u16*)carve((size_t)GN * GH * GGRIDS * 2);
    u16* wbf    = (u16*)carve((size_t)2 * 1024 * 1024 * 2);
    u16* abf    = (u16*)carve((size_t)GN * GEH * 2);
    u16* bbf    = (u16*)carve((size_t)GN * GEH * 2);
    float* bigf = (float*)carve((size_t)GN * GN * 4);      // dense adj, later logits
    float* cs   = (float*)carve((size_t)GN * 4);
    float* xw0  = (float*)carve((size_t)GN * GH * 4);
    float* xn1  = (float*)carve((size_t)GN * GH * 4);
    float* xw1  = (float*)carve((size_t)GN * GH * 4);
    float* xn2  = (float*)carve((size_t)GN * GH * 4);
    float* xp   = (float*)carve((size_t)GN * GH * 4);
    float* gbuf = (float*)carve((size_t)GN * GH * 4);
    float* xg   = (float*)carve((size_t)GN * GH * 4);
    float* xm   = (float*)carve((size_t)GN * GH * 4);
    float* agg  = (float*)carve((size_t)GN * GH * 4);
    float* deg  = (float*)carve((size_t)GN * 4);
    float* hm   = (float*)carve((size_t)GN * GH * 4);
    float* hl   = (float*)carve((size_t)GN * GH * 4);
    float* hbig = (float*)carve((size_t)GN * GEH * 4);
    float* hcv  = (float*)carve((size_t)GN * GEH * 4);
    float* ybuf = (float*)carve((size_t)GN * GEH * 4);
    float* outb = (float*)carve((size_t)GN * GH * 4);
    float* gt2  = (float*)carve((size_t)GN * GH * 4);
    float* bc   = (float*)carve((size_t)GEH * 4);
    float* zt   = (float*)carve((size_t)GN * GH2 * 4);
    float* zt2  = (float*)carve((size_t)GN * GH2 * 4);
    float* zs   = (float*)carve((size_t)GN * GH * 4);
    float* z1n  = (float*)carve((size_t)GN * GH * 4);
    float* z2n  = (float*)carve((size_t)GN * GH * 4);
    float* bmean = (float*)carve((size_t)GH2 * 4);
    float* brstd = (float*)carve((size_t)GH2 * 4);
    float* closs = (float*)carve(256);
    float* fw    = (float*)carve(256);
    float* xf   = (float*)carve((size_t)GN * GH * 4);
    float* xl   = (float*)carve((size_t)GN * GH * 4);
    float* xk   = (float*)carve((size_t)GN * GOUT * 4);
    float* t2k  = (float*)carve((size_t)GNUMR * GOUT * 4);
    float* cmean = (float*)carve((size_t)GOUT * 4);
    float* score = (float*)d_out;

    auto castb = [&](const float* s, u16* d, size_t n) {
        k_cast_bf16<<<ew(n), 256, 0, stream>>>(s, d, n);
    };
    // B operands are consumed transposed: src [K,N] -> dst [N,K]
    auto castT = [&](const float* s, u16* d, int R, int Cc) {
        k_cast_bf16_t<<<ew((size_t)R * Cc), 256, 0, stream>>>(s, d, R, Cc);
    };

    // ================= GCN branch =================
    // xw0 = x @ gcn0
    castb(x_in, bigbf, (size_t)GN * GIN);
    castT(gcn0, wbf, GIN, GH);
    gemm(stream, bigbf, wbf, nullptr, xw0, GN, GH, GIN, 0, nullptr);
    // dense normalized adjacency Ahat
    k_fillf<<<ew((size_t)GN * GN), 256, 0, stream>>>(bigf, 0.f, (size_t)GN * GN);
    k_scatter_edge<<<ew(GE), 256, 0, stream>>>(bigf, idx, attr, GE, GN);
    k_add_eye<<<ew(GN), 256, 0, stream>>>(bigf, GN);
    k_colsum<<<GN, 256, 0, stream>>>(bigf, cs, GN);
    k_norm_adj<<<ew((size_t)GN * GN), 256, 0, stream>>>(bigf, cs, Ahat, GN);
    // xn1 = relu(Ahat @ xw0)
    castT(xw0, bbf, GN, GH);
    gemm(stream, Ahat, bbf, nullptr, xn1, GN, GH, GN, GF_RELU, nullptr);
    // xn2 = relu(Ahat @ (xn1 @ gcn1))
    castb(xn1, abf, (size_t)GN * GH);
    castT(gcn1, wbf, GH, GH);
    gemm(stream, abf, wbf, nullptr, xw1, GN, GH, GH, 0, nullptr);
    castT(xw1, bbf, GN, GH);
    gemm(stream, Ahat, bbf, nullptr, xn2, GN, GH, GN, GF_RELU, nullptr);
    // xp = xn1 @ res_w + res_b   (abf still holds bf16(xn1))
    castT(res_w, wbf, GH, GH);
    gemm(stream, abf, wbf, res_b, xp, GN, GH, GH, GF_BIAS, nullptr);
    // g = sigmoid(cat(xp, xn2) @ gate_w + gate_b), split over the K-dim halves
    castb(xp, abf, (size_t)GN * GH);
    castT(gate_w, wbf, GH, GH);                       // rows 0..511
    gemm(stream, abf, wbf, nullptr, gbuf, GN, GH, GH, 0, nullptr);
    castb(xn2, abf, (size_t)GN * GH);
    castT(gate_w + (size_t)GH * GH, wbf, GH, GH);     // rows 512..1023
    gemm(stream, abf, wbf, gate_b, gbuf, GN, GH, GH, GF_ACCUM | GF_BIAS | GF_SIGMOID, nullptr);
    k_gate_combine<<<ew((size_t)GN * GH), 256, 0, stream>>>(xg, gbuf, xn2, xp, (size_t)GN * GH, 0);

    // ================= Mamba branch =================
    castb(adj_in, bigbf, (size_t)GN * GIN);
    castT(lin_w, wbf, GIN, GH);
    gemm(stream, bigbf, wbf, lin_b, xm, GN, GH, GIN, GF_BIAS, nullptr);

    for (int L = 0; L < 2; ++L) {
        int p = 12 + 16 * L;
        const float* ip_w = F(p + 0);  const float* ip_b = F(p + 1);
        const float* op_w = F(p + 2);  const float* op_b = F(p + 3);
        const float* cv_w = F(p + 4);  const float* cv_b = F(p + 5);
        const float* mB   = F(p + 7);  const float* mC   = F(p + 8);
        const float* mD   = F(p + 9);
        const float* gp_w = F(p + 10); const float* gp_b = F(p + 11);
        const float* lng  = F(p + 12); const float* lnb  = F(p + 13);
        const float* gt_w = F(p + 14); const float* gt_b = F(p + 15);

        // gate = sigmoid(xm @ gate_proj + b)
        castb(xm, abf, (size_t)GN * GH);
        castT(gt_w, wbf, GH, GH);
        gemm(stream, abf, wbf, gt_b, gt2, GN, GH, GH, GF_BIAS | GF_SIGMOID, nullptr);
        // graph aggregate
        k_fillf<<<ew((size_t)GN * GH), 256, 0, stream>>>(agg, 0.f, (size_t)GN * GH);
        k_fillf<<<ew(GN), 256, 0, stream>>>(deg, 0.f, GN);
        k_deg<<<ew(GE), 256, 0, stream>>>(deg, idx, GE);
        k_clampdeg<<<ew(GN), 256, 0, stream>>>(deg, GN);
        k_agg<<<GE, 128, 0, stream>>>(agg, xm, idx, GE);
        // h = xm + (agg/deg) @ graph_proj + b
        k_aggdiv_bf16<<<ew((size_t)GN * GH), 256, 0, stream>>>(agg, deg, abf, (size_t)GN * GH);
        k_copyf<<<ew((size_t)GN * GH), 256, 0, stream>>>(hm, xm, (size_t)GN * GH);
        castT(gp_w, wbf, GH, GH);
        gemm(stream, abf, wbf, gp_b, hm, GN, GH, GH, GF_BIAS | GF_ACCUM, nullptr);
        // layernorm
        k_layernorm<<<GN, 256, 0, stream>>>(hm, lng, lnb, hl, GH);
        // in_proj -> [N, EH]
        castb(hl, abf, (size_t)GN * GH);
        castT(ip_w, wbf, GH, GEH);
        gemm(stream, abf, wbf, ip_b, hbig, GN, GEH, GH, GF_BIAS, nullptr);
        // causal depthwise conv + SSM pointwise
        k_dwconv<<<ew((size_t)GN * GEH), 256, 0, stream>>>(hbig, cv_w, cv_b, hcv, GN, GEH);
        k_bc<<<ew(GEH), 256, 0, stream>>>(mB, mC, bc, GEH);
        k_ssm<<<ew((size_t)GN * GEH), 256, 0, stream>>>(hcv, bc, mD, ybuf, (size_t)GN * GEH);
        // out_proj
        castb(ybuf, abf, (size_t)GN * GEH);
        castT(op_w, wbf, GEH, GH);
        gemm(stream, abf, wbf, op_b, outb, GN, GH, GEH, GF_BIAS, nullptr);
        // xm = relu(gate*out + (1-gate)*xm)
        k_gate_combine<<<ew((size_t)GN * GH), 256, 0, stream>>>(xm, gt2, outb, xm, (size_t)GN * GH, 1);
    }

    // ================= projection heads + contrastive =================
    const float* zin[2] = { xg, xm };
    float* zout[2] = { z1n, z2n };
    for (int i = 0; i < 2; ++i) {
        castb(zin[i], abf, (size_t)GN * GH);
        castT(l1_w, wbf, GH, GH2);
        gemm(stream, abf, wbf, l1_b, zt, GN, GH2, GH, GF_BIAS, nullptr);
        k_colstats<<<GH2, 256, 0, stream>>>(zt, bmean, brstd, GN, GH2);
        k_bn_apply<<<ew((size_t)GN * GH2), 256, 0, stream>>>(zt, bmean, brstd, bn1g, bn1b, zt2,
                                                             (size_t)GN * GH2, GH2, 1);
        castb(zt2, abf, (size_t)GN * GH2);
        castT(l2_w, wbf, GH2, GH);
        gemm(stream, abf, wbf, l2_b, zs, GN, GH, GH2, GF_BIAS, nullptr);
        k_colstats<<<GH, 256, 0, stream>>>(zs, bmean, brstd, GN, GH);
        k_bn_apply<<<ew((size_t)GN * GH), 256, 0, stream>>>(zs, bmean, brstd, bn2g, bn2b, zs,
                                                            (size_t)GN * GH, GH, 0);
        k_rownorm<<<GN, 256, 0, stream>>>(zs, zout[i], GH);
    }
    // logits = (z1n @ z2n^T) / temp : Bt = z2n row-major [N,K] directly
    castb(z1n, abf, (size_t)GN * GH);
    castb(z2n, bbf, (size_t)GN * GH);
    gemm(stream, abf, bbf, nullptr, bigf, GN, GN, GH, GF_DIVSCALE, temp);
    k_fillf<<<1, 256, 0, stream>>>(closs, 0.f, 1);
    k_lsm_diag_row<<<GN, 256, 0, stream>>>(bigf, closs, GN);
    k_lsm_diag_col<<<GN, 256, 0, stream>>>(bigf, closs, GN);

    // ================= fusion =================
    castb(xg, abf, (size_t)GN * GH);
    castT(fus_w, wbf, GH, GH);
    gemm(stream, abf, wbf, nullptr, gbuf, GN, GH, GH, 0, nullptr);
    castb(xm, abf, (size_t)GN * GH);
    castT(fus_w + (size_t)GH * GH, wbf, GH, GH);
    gemm(stream, abf, wbf, fus_b, gbuf, GN, GH, GH, GF_ACCUM | GF_BIAS | GF_SIGMOID, nullptr);
    k_fusion_w<<<1, 64, 0, stream>>>(fus_dw, fw);
    k_fuse<<<ew((size_t)GN * GH), 256, 0, stream>>>(xg, xm, gbuf, fw, xf, (size_t)GN * GH);

    // ================= FastKAN + decoder =================
    k_layernorm<<<GN, 256, 0, stream>>>(xf, kan_g, kan_b, xl, GH);
    k_basis<<<ew((size_t)GN * GH), 256, 0, stream>>>(xl, kan_grid, basisb, GN);
    castT(spline_w, wbf, GH * GGRIDS, GOUT);
    gemm(stream, basisb, wbf, nullptr, xk, GN, GOUT, GH * GGRIDS, 0, nullptr);
    k_silu_bf16<<<ew((size_t)GN * GH), 256, 0, stream>>>(xf, abf, (size_t)GN * GH);
    castT(base_w, wbf, GH, GOUT);
    gemm(stream, abf, wbf, base_b, xk, GN, GOUT, GH, GF_ACCUM | GF_BIAS, nullptr);
    // mean-subtract over rows
    k_colmean<<<GOUT, 256, 0, stream>>>(xk, cmean, GN, GOUT);
    k_submean<<<ew((size_t)GN * GOUT), 256, 0, stream>>>(xk, cmean, (size_t)GN * GOUT, GOUT);
    // score = (xk[:NR] @ dec_w) @ xk[NR:]^T : Bt = xk[NR:] row-major [NR,K] directly
    castb(xk, abf, (size_t)GNUMR * GOUT);
    castT(dec_w, wbf, GOUT, GOUT);
    gemm(stream, abf, wbf, nullptr, t2k, GNUMR, GOUT, GOUT, 0, nullptr);
    castb(t2k, abf, (size_t)GNUMR * GOUT);
    castb(xk + (size_t)GNUMR * GOUT, bbf, (size_t)GNUMR * GOUT);
    gemm(stream, abf, bbf, nullptr, score, GNUMR, GNUMR, GOUT, 0, nullptr);
    // append closs
    k_put_scalar<<<1, 64, 0, stream>>>(score + (size_t)GNUMR * GNUMR, closs);
}
